// Convpass_swin_hypernet_mask_serial_22359599743398
// MI455X (gfx1250) — matrix operand order repaired and checked by reading, not compiled, not run on hardware
//
#include <hip/hip_runtime.h>
#include <hip/hip_bf16.h>

// ---------------------------------------------------------------------------
// CDNA5 (gfx1250) implementation: Convpass-swin hypernet adapter
// All GEMM stages on v_wmma_f32_16x16x32_f16 (wave32 WMMA).
// Weights pre-swizzled to WMMA-B fragment-major layout -> ds_load_b128 pairs.
// Conv halo staging via global_load_async_to_lds_b128 + s_wait_asynccnt.
// ---------------------------------------------------------------------------

typedef __attribute__((ext_vector_type(16))) _Float16 v16h;
typedef __attribute__((ext_vector_type(8)))  _Float16 v8h;
typedef __attribute__((ext_vector_type(8)))  float    v8f;

__device__ __forceinline__ v8f wmma_f16(v16h a, v16h b, v8f c) {
  return __builtin_amdgcn_wmma_f32_16x16x32_f16(false, a, false, b, (short)0, c,
                                                false, false);
}

__device__ __forceinline__ float qgelu(float v) {
  return v / (1.f + __expf(-1.702f * v));   // x * sigmoid(1.702 x)
}

__device__ __forceinline__ v16h cat8(v8h lo, v8h hi) {
  return __builtin_shufflevector(lo, hi, 0, 1, 2, 3, 4, 5, 6, 7,
                                         8, 9, 10, 11, 12, 13, 14, 15);
}

// A operand (16x32 f16) from a row-major f16 tile; this lane's row base.
// Elements: k = 16*(e>>3) + 8*(lane>>4) + (e&7)  -> two contiguous 8-half runs.
__device__ __forceinline__ v16h load_A16(const _Float16* row_base, int lane) {
  int khi = (lane >> 4) * 8;
  v8h lo = *(const v8h*)(row_base + khi);
  v8h hi = *(const v8h*)(row_base + 16 + khi);
  return cat8(lo, hi);
}

// A operand from fp32 global row (row_k = row_ptr + kt*32): float4 x4 + cvt.
__device__ __forceinline__ v16h load_A_f32(const float* rk, int khi) {
  float4 f0 = *(const float4*)(rk + khi);
  float4 f1 = *(const float4*)(rk + khi + 4);
  float4 f2 = *(const float4*)(rk + 16 + khi);
  float4 f3 = *(const float4*)(rk + 20 + khi);
  v16h a;
  a[0] = (_Float16)f0.x;  a[1] = (_Float16)f0.y;
  a[2] = (_Float16)f0.z;  a[3] = (_Float16)f0.w;
  a[4] = (_Float16)f1.x;  a[5] = (_Float16)f1.y;
  a[6] = (_Float16)f1.z;  a[7] = (_Float16)f1.w;
  a[8] = (_Float16)f2.x;  a[9] = (_Float16)f2.y;
  a[10] = (_Float16)f2.z; a[11] = (_Float16)f2.w;
  a[12] = (_Float16)f3.x; a[13] = (_Float16)f3.y;
  a[14] = (_Float16)f3.z; a[15] = (_Float16)f3.w;
  return a;
}

// Stage B = W^T (W row-major [N][K], row stride ld) into fragment-major LDS:
// dst[((kt*NT+nt)*32 + lane)*16 + e] = W[n][k],
//   k = kt*32 + 16*(e>>3) + 8*(lane>>4) + (e&7),  n = nt*16 + (lane&15)      (ISA 7.12.2)
template <int KT, int NT>
__device__ __forceinline__ void stage_BT(const float* __restrict__ W, int ld,
                                         _Float16* dst, int t, int nthreads) {
  const int total = KT * NT * 512;
  for (int i = t; i < total; i += nthreads) {
    int e = i & 15, lane = (i >> 4) & 31, f = i >> 9;
    int kt = f / NT, nt = f - kt * NT;
    int k = kt * 32 + ((e >> 3) << 4) + ((lane >> 4) << 3) + (e & 7);
    int n = nt * 16 + (lane & 15);
    dst[i] = (_Float16)W[n * ld + k];
  }
}

#define NPOS 100352l   // 32*56*56

// ---------------------------------------------------------------------------
// K1: pf = relu(x @ W1^T + b1) @ W2^T + b2     [N,64] fp32 -> ws
// One wave = one 16-row tile; 8 waves/block; 784 blocks = 6272 tiles.
// ---------------------------------------------------------------------------
__global__ __launch_bounds__(256) void k_meta_pf(const float* __restrict__ x,
                                                 const float* __restrict__ W1,
                                                 const float* __restrict__ b1,
                                                 const float* __restrict__ W2,
                                                 const float* __restrict__ b2,
                                                 float* __restrict__ pf) {
  __shared__ _Float16 sW1f[6 * 4 * 512];   // W1^T fragments (24 KB)
  __shared__ _Float16 sW2f[2 * 4 * 512];   // W2^T fragments (8 KB)
  __shared__ _Float16 sPF1[8][16 * 64];    // per-wave relu(layer1) tile (16 KB)
  int t = threadIdx.x;
  stage_BT<6, 4>(W1, 192, sW1f, t, 256);
  stage_BT<2, 4>(W2, 64, sW2f, t, 256);
  __syncthreads();

  int wave = t >> 5, lane = t & 31;
  long row0 = ((long)blockIdx.x * 8 + wave) * 16;
  int arow = lane & 15, khi = (lane >> 4) * 8;
  const float* xr = x + (row0 + arow) * 192;

  v8f acc[4] = {};
  for (int kt = 0; kt < 6; ++kt) {           // K = 192
    v16h a = load_A_f32(xr + kt * 32, khi);
    const _Float16* bf = sW1f + (kt * 4) * 512 + (lane << 4);
#pragma unroll
    for (int nt = 0; nt < 4; ++nt)
      acc[nt] = wmma_f16(a, *(const v16h*)(bf + (nt << 9)), acc[nt]);
  }
  // bias + relu -> per-wave LDS tile (f16)
  _Float16* pfw = sPF1[wave];
#pragma unroll
  for (int nt = 0; nt < 4; ++nt)
#pragma unroll
    for (int r = 0; r < 8; ++r) {
      int m = r + (lane >> 4) * 8, n = nt * 16 + (lane & 15);
      float v = acc[nt][r] + b1[n];
      pfw[m * 64 + n] = (_Float16)(v > 0.f ? v : 0.f);
    }
  __syncthreads();

  v8f acc2[4] = {};
  for (int kt = 0; kt < 2; ++kt) {           // K = 64
    v16h a = load_A16(pfw + (lane & 15) * 64 + kt * 32, lane);
    const _Float16* bf = sW2f + (kt * 4) * 512 + (lane << 4);
#pragma unroll
    for (int nt = 0; nt < 4; ++nt)
      acc2[nt] = wmma_f16(a, *(const v16h*)(bf + (nt << 9)), acc2[nt]);
  }
#pragma unroll
  for (int nt = 0; nt < 4; ++nt)
#pragma unroll
    for (int r = 0; r < 8; ++r) {
      int m = r + (lane >> 4) * 8, n = nt * 16 + (lane & 15);
      pf[(row0 + m) * 64 + n] = acc2[nt][r] + b2[n];
    }
}

// ---------------------------------------------------------------------------
// K2: per-sample softmax routing + mwf reduction. One block per batch sample.
// ---------------------------------------------------------------------------
__global__ __launch_bounds__(256) void k_mask_mwf(const float* __restrict__ pf,
                                                  const float* __restrict__ mtok,
                                                  float* __restrict__ Lbuf,
                                                  float* __restrict__ wbuf,
                                                  float* __restrict__ mwf) {
  int b = blockIdx.x, t = threadIdx.x;
  __shared__ float stok[8 * 64];
  __shared__ float sred[256];
  __shared__ float smwf[4][64];
  for (int i = t; i < 512; i += 256) stok[i] = mtok[i];
  __syncthreads();

  const float* pfb = pf + (long)b * 3136 * 64;
  float* Lb = Lbuf + (long)b * 8 * 3136;

  float mmax[8];
#pragma unroll
  for (int m = 0; m < 8; ++m) mmax[m] = -1e30f;
  for (int p = t; p < 3136; p += 256) {
    const float* v = pfb + p * 64;
    float l[8] = {};
    for (int k = 0; k < 64; ++k) {
      float xv = v[k];
#pragma unroll
      for (int m = 0; m < 8; ++m) l[m] += xv * stok[m * 64 + k];
    }
#pragma unroll
    for (int m = 0; m < 8; ++m) {
      Lb[m * 3136 + p] = l[m];
      mmax[m] = fmaxf(mmax[m], l[m]);
    }
  }
  for (int m = 0; m < 8; ++m) {            // block max
    sred[t] = mmax[m]; __syncthreads();
    for (int s = 128; s > 0; s >>= 1) {
      if (t < s) sred[t] = fmaxf(sred[t], sred[t + s]);
      __syncthreads();
    }
    mmax[m] = sred[0]; __syncthreads();
  }
  float msum[8] = {};
  for (int p = t; p < 3136; p += 256)
#pragma unroll
    for (int m = 0; m < 8; ++m) msum[m] += __expf(Lb[m * 3136 + p] - mmax[m]);
  for (int m = 0; m < 8; ++m) {            // block sum
    sred[t] = msum[m]; __syncthreads();
    for (int s = 128; s > 0; s >>= 1) {
      if (t < s) sred[t] += sred[t + s];
      __syncthreads();
    }
    msum[m] = sred[0]; __syncthreads();
  }
  float* wb = wbuf + (long)b * 3136;
  for (int p = t; p < 3136; p += 256) {
    float w = 0.f;
#pragma unroll
    for (int m = 0; m < 8; ++m) w += __expf(Lb[m * 3136 + p] - mmax[m]) / msum[m];
    wb[p] = w;
  }
  __syncthreads();
  int j = t & 63, g = t >> 6;
  float a = 0.f;
  for (int p = g; p < 3136; p += 4) a += wb[p] * pfb[p * 64 + j];
  smwf[g][j] = a;
  __syncthreads();
  if (t < 64) mwf[b * 64 + t] = smwf[0][t] + smwf[1][t] + smwf[2][t] + smwf[3][t];
}

// ---------------------------------------------------------------------------
// K3: hypernet -> per-sample conv weights, emitted DIRECTLY in WMMA-B
// fragment-major f16 layout: frag (kc 0..17, nt 0..3), lane, e.
// B-k index kk = tap*64 + din (tap = ky*3+kx), n = dout.
// ---------------------------------------------------------------------------
__global__ __launch_bounds__(256) void k_hyper(const float* __restrict__ mwf,
                                               const float* __restrict__ Wh,
                                               const float* __restrict__ bh,
                                               _Float16* __restrict__ convwh) {
  long o = (long)blockIdx.x * 256 + threadIdx.x;
  if (o >= 32l * 36864) return;
  int b = (int)(o / 36864), k = (int)(o % 36864);   // k = ((dout*64+din)*9+tap)
  int dout = k / (64 * 9), rem = k % (64 * 9), din = rem / 9, tap = rem % 9;
  const float* mv = mwf + b * 64;
  const float* wr = Wh + (long)k * 64;
  float s = bh[k];
  for (int q = 0; q < 64; ++q) s += mv[q] * wr[q];
  int kk = tap * 64 + din;
  int kc = kk >> 5, r5 = kk & 31, e = ((r5 >> 4) << 3) | (r5 & 7), lh = (r5 >> 3) & 1;
  int lane = lh * 16 + (dout & 15), nt = dout >> 4;
  convwh[(((long)b * 72 + kc * 4 + nt) << 9) + (lane << 4) + e] = (_Float16)s;
}

// ---------------------------------------------------------------------------
// K4: adapter down: xp = quick_gelu(x @ Wd^T + bd), NHWC f16  [N,64]
// ---------------------------------------------------------------------------
__global__ __launch_bounds__(256) void k_down(const float* __restrict__ x,
                                              const float* __restrict__ Wd,
                                              const float* __restrict__ bd,
                                              _Float16* __restrict__ xp) {
  __shared__ _Float16 sWf[6 * 4 * 512];   // Wd^T fragments (24 KB)
  int t = threadIdx.x;
  stage_BT<6, 4>(Wd, 192, sWf, t, 256);
  __syncthreads();

  int wave = t >> 5, lane = t & 31;
  long row0 = ((long)blockIdx.x * 8 + wave) * 16;
  int arow = lane & 15, khi = (lane >> 4) * 8;
  const float* xr = x + (row0 + arow) * 192;
  v8f acc[4] = {};
  for (int kt = 0; kt < 6; ++kt) {
    v16h a = load_A_f32(xr + kt * 32, khi);
    const _Float16* bf = sWf + (kt * 4) * 512 + (lane << 4);
#pragma unroll
    for (int nt = 0; nt < 4; ++nt)
      acc[nt] = wmma_f16(a, *(const v16h*)(bf + (nt << 9)), acc[nt]);
  }
#pragma unroll
  for (int nt = 0; nt < 4; ++nt)
#pragma unroll
    for (int r = 0; r < 8; ++r) {
      int m = r + (lane >> 4) * 8, n = nt * 16 + (lane & 15);
      xp[(row0 + m) * 64 + n] = (_Float16)qgelu(acc[nt][r] + bd[n]);
    }
}

// ---------------------------------------------------------------------------
// K5: per-sample 3x3 conv as im2col GEMM (K = 9*64 = 576) + quick_gelu.
// Block = one (b, y) image row; halo rows staged with async LDS loads.
// A K-chunk (32) never crosses a tap boundary -> A = two ds_load_b128.
// B fragments are contiguous per lane in global (k_hyper layout).
// ---------------------------------------------------------------------------
__global__ __launch_bounds__(128) void k_conv(const _Float16* __restrict__ xp,
                                              const _Float16* __restrict__ convwh,
                                              _Float16* __restrict__ xg) {
  __shared__ _Float16 sX[3 * 74 * 64];   // [ky][px 0..73][din]; px0 == x=-1
  int t = threadIdx.x;
  int b = blockIdx.x / 56, y = blockIdx.x % 56;
  for (int i = t; i < 3 * 74 * 64; i += 128) sX[i] = (_Float16)0.f;
  __syncthreads();

  // async halo stage: 3 rows x 56 px x 64 ch (f16) = 3*448 b128 chunks
  for (int i = t; i < 3 * 448; i += 128) {
    int ky = i / 448, c = i % 448;
    int yy = y + ky - 1;
    if (yy < 0 || yy >= 56) continue;
    const _Float16* src = xp + (((long)b * 56 + yy) * 56) * 64 + c * 8;
    unsigned dst = (unsigned)(uintptr_t)&sX[(ky * 74 + 1) * 64 + c * 8];
    asm volatile("global_load_async_to_lds_b128 %0, %1, off"
                 :: "v"(dst), "v"((unsigned long long)(uintptr_t)src)
                 : "memory");
  }
  asm volatile("s_wait_asynccnt 0" ::: "memory");
  __syncthreads();

  int wave = t >> 5, lane = t & 31;
  int x0 = wave * 16;
  int pix = lane & 15, khi = (lane >> 4) * 8;
  const _Float16* wfrag = convwh + (((long)b * 72) << 9) + (lane << 4);

  v8f acc[4] = {};
  for (int kc = 0; kc < 18; ++kc) {          // K = 576, tap = kc>>1 fixed/chunk
    int tap = kc >> 1;
    int ky = tap / 3, kx = tap % 3;
    int dbase = (kc & 1) * 32 + khi;
    const _Float16* ap = &sX[(ky * 74 + (x0 + pix + kx)) * 64];
    v8h lo = *(const v8h*)(ap + dbase);
    v8h hi = *(const v8h*)(ap + dbase + 16);
    v16h a = cat8(lo, hi);
    const _Float16* bf = wfrag + ((kc * 4) << 9);
#pragma unroll
    for (int nt = 0; nt < 4; ++nt)
      acc[nt] = wmma_f16(a, *(const v16h*)(bf + (nt << 9)), acc[nt]);
  }
#pragma unroll
  for (int nt = 0; nt < 4; ++nt)
#pragma unroll
    for (int r = 0; r < 8; ++r) {
      int m = r + (lane >> 4) * 8;
      int xc = x0 + m;
      if (xc < 56)
        xg[(((long)b * 56 + y) * 56 + xc) * 64 + nt * 16 + (lane & 15)] =
            (_Float16)qgelu(acc[nt][r]);
    }
}

// ---------------------------------------------------------------------------
// K6: adapter up + bias + residual: out = xg @ Wu^T + bu + x   [N,192] fp32
// ---------------------------------------------------------------------------
__global__ __launch_bounds__(256) void k_up(const _Float16* __restrict__ xg,
                                            const float* __restrict__ Wu,
                                            const float* __restrict__ bu,
                                            const float* __restrict__ x,
                                            float* __restrict__ out) {
  __shared__ _Float16 sWf[2 * 12 * 512];  // Wu^T fragments (24 KB)
  int t = threadIdx.x;
  stage_BT<2, 12>(Wu, 64, sWf, t, 256);
  __syncthreads();

  int wave = t >> 5, lane = t & 31;
  long row0 = ((long)blockIdx.x * 8 + wave) * 16;
  __builtin_prefetch(x + row0 * 192, 0, 1);
  int arow = lane & 15;
  const _Float16* ar = xg + (row0 + arow) * 64;
  v16h a0 = load_A16(ar, lane);
  v16h a1 = load_A16(ar + 32, lane);

  v8f acc[12] = {};
#pragma unroll
  for (int nt = 0; nt < 12; ++nt) {
    acc[nt] = wmma_f16(a0, *(const v16h*)(sWf + ((0 * 12 + nt) << 9) + (lane << 4)), acc[nt]);
    acc[nt] = wmma_f16(a1, *(const v16h*)(sWf + ((1 * 12 + nt) << 9) + (lane << 4)), acc[nt]);
  }

#pragma unroll
  for (int nt = 0; nt < 12; ++nt)
#pragma unroll
    for (int r = 0; r < 8; ++r) {
      int m = r + (lane >> 4) * 8, n = nt * 16 + (lane & 15);
      long idx = (row0 + m) * 192 + n;
      out[idx] = acc[nt][r] + bu[n] + x[idx];
    }
}

// ---------------------------------------------------------------------------
extern "C" void kernel_launch(void* const* d_in, const int* in_sizes, int n_in,
                              void* d_out, int out_size, void* d_ws, size_t ws_size,
                              hipStream_t stream) {
  (void)in_sizes; (void)n_in; (void)out_size; (void)ws_size;
  const float* x    = (const float*)d_in[0];
  const float* W1   = (const float*)d_in[1];
  const float* b1   = (const float*)d_in[2];
  const float* W2   = (const float*)d_in[3];
  const float* b2   = (const float*)d_in[4];
  const float* mtok = (const float*)d_in[5];
  const float* Wh   = (const float*)d_in[6];
  const float* bh   = (const float*)d_in[7];
  const float* Wd   = (const float*)d_in[8];
  const float* bd   = (const float*)d_in[9];
  const float* Wu   = (const float*)d_in[10];
  const float* bu   = (const float*)d_in[11];
  float* out = (float*)d_out;

  float* pf   = (float*)d_ws;                    // N*64
  float* Lbuf = pf + NPOS * 64;                  // 32*8*3136
  float* wbuf = Lbuf + 32l * 8 * 3136;           // N
  float* mwf  = wbuf + NPOS;                     // 32*64
  _Float16* convwh = (_Float16*)(mwf + 32 * 64); // 32*72*512
  _Float16* xp = convwh + 32l * 72 * 512;        // N*64
  _Float16* xg = xp + NPOS * 64;                 // N*64

  k_meta_pf<<<784, 256, 0, stream>>>(x, W1, b1, W2, b2, pf);
  k_mask_mwf<<<32, 256, 0, stream>>>(pf, mtok, Lbuf, wbuf, mwf);
  k_hyper<<<(int)((32l * 36864 + 255) / 256), 256, 0, stream>>>(mwf, Wh, bh, convwh);
  k_down<<<784, 256, 0, stream>>>(x, Wd, bd, xp);
  k_conv<<<32 * 56, 128, 0, stream>>>(xp, convwh, xg);
  k_up<<<784, 256, 0, stream>>>(xg, Wu, bu, x, out);
}
